// MultiHeadAttentionFast_53042846105897
// MI455X (gfx1250) — matrix operand order repaired and checked by reading, not compiled, not run on hardware
//
#include <hip/hip_runtime.h>

#define BSZ   2
#define SEQ   2048
#define EMB   1024
#define NH    16
#define HD    64
#define QKVN  (3 * EMB)

typedef __attribute__((ext_vector_type(16))) _Float16 v16h;
typedef __attribute__((ext_vector_type(8)))  _Float16 v8h;
typedef __attribute__((ext_vector_type(8)))  float    v8f;
typedef __attribute__((ext_vector_type(4)))  float    v4f;

// D = A*B + C, 16x16x32 f16 -> f32 accum
__device__ __forceinline__ v8f wmma16(v16h a, v16h b, v8f c) {
  return __builtin_amdgcn_wmma_f32_16x16x32_f16(
      /*neg_a=*/false, a, /*neg_b=*/false, b,
      /*c_mod=*/(short)0, c, /*reuse_a=*/false, /*reuse_b=*/false);
}

// Build a 16-element f16 fragment from two contiguous 8-element (16B) LDS chunks.
__device__ __forceinline__ v16h frag2(const _Float16* lo, const _Float16* hi) {
  v8h a = *(const v8h*)lo;
  v8h b = *(const v8h*)hi;
  v16h r;
#pragma unroll
  for (int i = 0; i < 8; ++i) { r[i] = a[i]; r[i + 8] = b[i]; }
  return r;
}

// CDNA5 async memory->LDS copy, 16B per lane, tracked by ASYNCcnt.
// LDS operand = low 32 bits of the generic shared-pointer (ISA 10.2).
__device__ __forceinline__ void async_copy16(const _Float16* gp, _Float16* lp) {
  unsigned int loff = (unsigned int)(unsigned long long)lp;
  asm volatile("global_load_async_to_lds_b128 %0, %1, off"
               :: "v"(loff), "v"(gp) : "memory");
}
__device__ __forceinline__ void async_wait0() {
  asm volatile("s_wait_asynccnt 0x0" ::: "memory");
}

// ---------------------------------------------------------------------------
// Kernel 1: QKV projection.  C[4096,3072] = X[4096,1024] @ Wqkv + bqkv,
// scattered as f16 into Q/K/V buffers laid out [B*H][SEQ][HD].
// Block: 256 thr (8 waves). Block tile 128(M) x 64(N), K-chunk 32.
// Double-buffered LDS; global loads batched into regs (one wait per tile).
// ---------------------------------------------------------------------------
__global__ void __launch_bounds__(256)
qkv_gemm_kernel(const float* __restrict__ X, const float* __restrict__ W,
                const float* __restrict__ bias,
                _Float16* __restrict__ Qb, _Float16* __restrict__ Kb,
                _Float16* __restrict__ Vb) {
  __shared__ _Float16 As[2][128][40];   // row-major A tile (pad keeps 16B align)
  __shared__ _Float16 Bt[2][64][40];    // B tile transposed: Bt[n][k]

  const int tid   = threadIdx.x;
  const int lane  = tid & 31;
  const int wave  = tid >> 5;
  const int m0    = blockIdx.y * 128;
  const int n0    = blockIdx.x * 64;
  const int rbase = wave * 16;
  const int col   = lane & 15;
  const int hiOff = (lane >> 4) * 8;
  const int arow  = rbase + col;

  v4f aw[4], bw[2];
  auto loadTile = [&](int kk) {
#pragma unroll
    for (int it = 0; it < 4; ++it) {
      int e = it * 1024 + tid * 4;
      aw[it] = *(const v4f*)(X + (size_t)(m0 + (e >> 5)) * EMB + kk + (e & 31));
    }
#pragma unroll
    for (int it = 0; it < 2; ++it) {
      int e = it * 1024 + tid * 4;
      bw[it] = *(const v4f*)(W + (size_t)(kk + (e >> 6)) * QKVN + n0 + (e & 63));
    }
  };
  auto storeTile = [&](int buf) {
#pragma unroll
    for (int it = 0; it < 4; ++it) {
      int e = it * 1024 + tid * 4;
      int r = e >> 5, c = e & 31;
#pragma unroll
      for (int u = 0; u < 4; ++u) As[buf][r][c + u] = (_Float16)aw[it][u];
    }
#pragma unroll
    for (int it = 0; it < 2; ++it) {
      int e = it * 1024 + tid * 4;
      int r = e >> 6, c = e & 63;
#pragma unroll
      for (int u = 0; u < 4; ++u) Bt[buf][c + u][r] = (_Float16)bw[it][u];
    }
  };

  v8f vz = {};
  v8f acc[4];
#pragma unroll
  for (int j = 0; j < 4; ++j) acc[j] = vz;

  loadTile(0);
  storeTile(0);
  int cur = 0;
  for (int kk = 0; kk < EMB; kk += 32) {
    __syncthreads();
    const bool more = (kk + 32 < EMB);
    if (more) loadTile(kk + 32);            // hide global latency under WMMAs

    v16h a = frag2(&As[cur][arow][hiOff], &As[cur][arow][hiOff + 16]);
#pragma unroll
    for (int j = 0; j < 4; ++j) {
      const _Float16* bp = &Bt[cur][j * 16 + col][hiOff];
      acc[j] = wmma16(a, frag2(bp, bp + 16), acc[j]);
    }
    if (more) { storeTile(cur ^ 1); cur ^= 1; }
  }

  // epilogue: bias + scatter f16 into Q/K/V [B*H][SEQ][HD]
#pragma unroll
  for (int j = 0; j < 4; ++j) {
#pragma unroll
    for (int r = 0; r < 8; ++r) {
      int m = m0 + rbase + r + hiOff;
      int n = n0 + j * 16 + col;
      float v = acc[j][r] + bias[n];
      int b = m >> 11, s = m & (SEQ - 1);
      int which = n >> 10;               // 0=Q 1=K 2=V
      int e = n & (EMB - 1);
      int h = e >> 6, d = e & (HD - 1);
      size_t off = (((size_t)(b * NH + h)) * SEQ + s) * HD + d;
      _Float16 hv = (_Float16)v;
      if (which == 0)      Qb[off] = hv;
      else if (which == 1) Kb[off] = hv;
      else                 Vb[off] = hv;
    }
  }
}

// ---------------------------------------------------------------------------
// Kernel 2: causal flash attention over f16 Q/K/V, f32 context out.
// Block: 128 thr (4 waves), one block per (head, 64-query tile).
// Q and K staged via CDNA5 async global->LDS copies (ASYNCcnt).
// ---------------------------------------------------------------------------
__global__ void __launch_bounds__(128)
attn_kernel(const _Float16* __restrict__ Qb, const _Float16* __restrict__ Kb,
            const _Float16* __restrict__ Vb, float* __restrict__ Ctx) {
  __shared__ _Float16 Qs[64][72];   // [q][d]
  __shared__ _Float16 Ks[64][72];   // [key][d]
  __shared__ _Float16 Vt[64][72];   // [d][key] (transposed V)
  __shared__ _Float16 Ps[64][72];   // [q][key] softmax probs

  const int tid   = threadIdx.x;
  const int lane  = tid & 31;
  const int wave  = tid >> 5;          // 0..3
  const int qt    = blockIdx.x;
  const int bh    = blockIdx.y;        // b*NH + h
  const int b     = bh >> 4, h = bh & 15;
  const int q0    = qt * 64;
  const int rbase = wave * 16;
  const int col   = lane & 15;
  const int hiOff = (lane >> 4) * 8;
  const size_t headBase = (size_t)bh * SEQ * HD;

  // stage Q once (async copy, 16B per lane per issue)
  {
    const _Float16* Qg = Qb + headBase + (size_t)q0 * HD;
#pragma unroll
    for (int it = 0; it < 4; ++it) {
      int e = it * 1024 + tid * 8;
      int r = e >> 6, c = e & 63;
      async_copy16(Qg + r * HD + c, &Qs[r][c]);
    }
  }

  v8f vz = {};
  v8f o[4];
#pragma unroll
  for (int j = 0; j < 4; ++j) o[j] = vz;
  float mrow[8], lrow[8];
#pragma unroll
  for (int r = 0; r < 8; ++r) { mrow[r] = -1e30f; lrow[r] = 0.0f; }

  for (int kt = 0; kt <= qt; ++kt) {
    const int k0 = kt * 64;
    __syncthreads();                        // prior iteration's LDS reads done
    const _Float16* Kg = Kb + headBase + (size_t)k0 * HD;
    const _Float16* Vg = Vb + headBase + (size_t)k0 * HD;
    // K: async straight into LDS
#pragma unroll
    for (int it = 0; it < 4; ++it) {
      int e = it * 1024 + tid * 8;
      async_copy16(Kg + (e >> 6) * HD + (e & 63), &Ks[e >> 6][e & 63]);
    }
    // V: batch loads into regs, then transpose-scatter to Vt
    v8h wv[4];
#pragma unroll
    for (int it = 0; it < 4; ++it) {
      int e = it * 1024 + tid * 8;
      wv[it] = *(const v8h*)(Vg + (e >> 6) * HD + (e & 63));
    }
#pragma unroll
    for (int it = 0; it < 4; ++it) {
      int e = it * 1024 + tid * 8;
      int r = e >> 6, c = e & 63;
#pragma unroll
      for (int u = 0; u < 8; ++u) Vt[c + u][r] = wv[it][u];
    }
    async_wait0();                          // K (and Q on first iter) landed
    __syncthreads();

    // S = Q K^T (16x64 per wave), contraction over d=64 in two 32-chunks
    v8f s[4];
#pragma unroll
    for (int j = 0; j < 4; ++j) s[j] = vz;
#pragma unroll
    for (int kc = 0; kc < 64; kc += 32) {
      const _Float16* qp = &Qs[rbase + col][kc + hiOff];
      v16h aq = frag2(qp, qp + 16);
#pragma unroll
      for (int j = 0; j < 4; ++j) {
        const _Float16* bp = &Ks[j * 16 + col][kc + hiOff];
        s[j] = wmma16(aq, frag2(bp, bp + 16), s[j]);
      }
    }

    // scale + causal mask + row max
    float rmax[8];
#pragma unroll
    for (int r = 0; r < 8; ++r) rmax[r] = -1e30f;
#pragma unroll
    for (int j = 0; j < 4; ++j) {
#pragma unroll
      for (int r = 0; r < 8; ++r) {
        float x = s[j][r] * 0.125f;                 // 1/sqrt(64)
        int kg = k0 + j * 16 + col;
        int qg = q0 + rbase + r + hiOff;
        if (kg > qg) x = -1e30f;
        s[j][r] = x;
        rmax[r] = fmaxf(rmax[r], x);
      }
    }
#pragma unroll
    for (int r = 0; r < 8; ++r)
#pragma unroll
      for (int msk = 1; msk < 16; msk <<= 1)
        rmax[r] = fmaxf(rmax[r], __shfl_xor(rmax[r], msk, 32));

    // online softmax update
    float corr[8], psum[8];
#pragma unroll
    for (int r = 0; r < 8; ++r) {
      float mn = fmaxf(mrow[r], rmax[r]);
      corr[r] = __expf(mrow[r] - mn);
      mrow[r] = mn;
      psum[r] = 0.0f;
    }
#pragma unroll
    for (int j = 0; j < 4; ++j)
#pragma unroll
      for (int r = 0; r < 8; ++r) {
        float p = __expf(s[j][r] - mrow[r]);
        psum[r] += p;
        Ps[rbase + r + hiOff][j * 16 + col] = (_Float16)p;
      }
#pragma unroll
    for (int r = 0; r < 8; ++r) {
#pragma unroll
      for (int msk = 1; msk < 16; msk <<= 1)
        psum[r] += __shfl_xor(psum[r], msk, 32);
      lrow[r] = lrow[r] * corr[r] + psum[r];
    }
#pragma unroll
    for (int j = 0; j < 4; ++j)
#pragma unroll
      for (int r = 0; r < 8; ++r) o[j][r] *= corr[r];
    __syncthreads();

    // O += P V, contraction over 64 keys in two 32-chunks
#pragma unroll
    for (int kc = 0; kc < 64; kc += 32) {
      const _Float16* pp = &Ps[rbase + col][kc + hiOff];
      v16h ap = frag2(pp, pp + 16);
#pragma unroll
      for (int j = 0; j < 4; ++j) {
        const _Float16* bp = &Vt[j * 16 + col][kc + hiOff];
        o[j] = wmma16(ap, frag2(bp, bp + 16), o[j]);
      }
    }
  }

  // finalize: divide by row sums, write f32 context [B][SEQ][EMB]
#pragma unroll
  for (int r = 0; r < 8; ++r) {
    float inv = 1.0f / lrow[r];
    int qg = q0 + rbase + r + hiOff;
#pragma unroll
    for (int j = 0; j < 4; ++j) {
      int d = j * 16 + col;
      Ctx[((size_t)(b * SEQ + qg)) * EMB + h * HD + d] = o[j][r] * inv;
    }
  }
}

// ---------------------------------------------------------------------------
// Kernel 3: output projection. Out[4096,1024] = Ctx @ Wout + bout (f32 out).
// Same double-buffered WMMA GEMM skeleton as kernel 1.
// ---------------------------------------------------------------------------
__global__ void __launch_bounds__(256)
out_proj_kernel(const float* __restrict__ X, const float* __restrict__ W,
                const float* __restrict__ bias, float* __restrict__ Out) {
  __shared__ _Float16 As[2][128][40];
  __shared__ _Float16 Bt[2][64][40];

  const int tid   = threadIdx.x;
  const int lane  = tid & 31;
  const int wave  = tid >> 5;
  const int m0    = blockIdx.y * 128;
  const int n0    = blockIdx.x * 64;
  const int rbase = wave * 16;
  const int col   = lane & 15;
  const int hiOff = (lane >> 4) * 8;
  const int arow  = rbase + col;

  v4f aw[4], bw[2];
  auto loadTile = [&](int kk) {
#pragma unroll
    for (int it = 0; it < 4; ++it) {
      int e = it * 1024 + tid * 4;
      aw[it] = *(const v4f*)(X + (size_t)(m0 + (e >> 5)) * EMB + kk + (e & 31));
    }
#pragma unroll
    for (int it = 0; it < 2; ++it) {
      int e = it * 1024 + tid * 4;
      bw[it] = *(const v4f*)(W + (size_t)(kk + (e >> 6)) * EMB + n0 + (e & 63));
    }
  };
  auto storeTile = [&](int buf) {
#pragma unroll
    for (int it = 0; it < 4; ++it) {
      int e = it * 1024 + tid * 4;
      int r = e >> 5, c = e & 31;
#pragma unroll
      for (int u = 0; u < 4; ++u) As[buf][r][c + u] = (_Float16)aw[it][u];
    }
#pragma unroll
    for (int it = 0; it < 2; ++it) {
      int e = it * 1024 + tid * 4;
      int r = e >> 6, c = e & 63;
#pragma unroll
      for (int u = 0; u < 4; ++u) Bt[buf][c + u][r] = (_Float16)bw[it][u];
    }
  };

  v8f vz = {};
  v8f acc[4];
#pragma unroll
  for (int j = 0; j < 4; ++j) acc[j] = vz;

  loadTile(0);
  storeTile(0);
  int cur = 0;
  for (int kk = 0; kk < EMB; kk += 32) {
    __syncthreads();
    const bool more = (kk + 32 < EMB);
    if (more) loadTile(kk + 32);

    v16h a = frag2(&As[cur][arow][hiOff], &As[cur][arow][hiOff + 16]);
#pragma unroll
    for (int j = 0; j < 4; ++j) {
      const _Float16* bp = &Bt[cur][j * 16 + col][hiOff];
      acc[j] = wmma16(a, frag2(bp, bp + 16), acc[j]);
    }
    if (more) { storeTile(cur ^ 1); cur ^= 1; }
  }

#pragma unroll
  for (int j = 0; j < 4; ++j) {
#pragma unroll
    for (int r = 0; r < 8; ++r) {
      int m = m0 + rbase + r + hiOff;
      int n = n0 + j * 16 + col;
      Out[(size_t)m * EMB + n] = acc[j][r] + bias[n];
    }
  }
}

// ---------------------------------------------------------------------------
extern "C" void kernel_launch(void* const* d_in, const int* in_sizes, int n_in,
                              void* d_out, int out_size, void* d_ws, size_t ws_size,
                              hipStream_t stream) {
  (void)in_sizes; (void)n_in; (void)out_size; (void)ws_size;
  const float* batch = (const float*)d_in[0];
  const float* Wqkv  = (const float*)d_in[1];
  const float* bqkv  = (const float*)d_in[2];
  const float* Wout  = (const float*)d_in[3];
  const float* bout  = (const float*)d_in[4];
  float* out = (float*)d_out;

  const size_t headElems = (size_t)BSZ * NH * SEQ * HD;   // 4,194,304
  _Float16* qb = (_Float16*)d_ws;
  _Float16* kb = qb + headElems;
  _Float16* vb = kb + headElems;
  float*   ctx = (float*)(vb + headElems);                // total ws = 40 MB

  dim3 g1(QKVN / 64, (BSZ * SEQ) / 128);                  // 48 x 32
  qkv_gemm_kernel<<<g1, 256, 0, stream>>>(batch, Wqkv, bqkv, qb, kb, vb);

  dim3 g2(SEQ / 64, BSZ * NH);                            // 32 x 32
  attn_kernel<<<g2, 128, 0, stream>>>(qb, kb, vb, ctx);

  dim3 g3(EMB / 64, (BSZ * SEQ) / 128);                   // 16 x 32
  out_proj_kernel<<<g3, 256, 0, stream>>>(ctx, Wout, bout, out);
}
